// AttentionBlock_8555574854505
// MI455X (gfx1250) — matrix operand order repaired and checked by reading, not compiled, not run on hardware
//
#include <hip/hip_runtime.h>

typedef __attribute__((ext_vector_type(16))) _Float16 v16h;
typedef __attribute__((ext_vector_type(8)))  _Float16 v8h;
typedef __attribute__((ext_vector_type(8)))  float    v8f;
typedef __attribute__((ext_vector_type(4)))  float    v4f;

#define D_IN  1024
#define DH    128
#define SEQ   4096
#define NBAT  4
#define BT    (NBAT * SEQ)
#define CHUNK 64   // K-dim staging chunk for projection GEMM

static __device__ __forceinline__ v8f wmma16(v16h a, v16h b, v8f c) {
  // D = A(16x32 f16) * B(32x16 f16) + C(16x16 f32)
  return __builtin_amdgcn_wmma_f32_16x16x32_f16(false, a, false, b, (short)0, c,
                                                false, false);
}

static __device__ __forceinline__ v8f vzero8() {
  v8f z = {0.f, 0.f, 0.f, 0.f, 0.f, 0.f, 0.f, 0.f};
  return z;
}

// 32-bit LDS byte offset of a __shared__ object (addrspacecast generic->AS3).
static __device__ __forceinline__ unsigned lds_addr(const void* p) {
  typedef __attribute__((address_space(3))) const void* lds_ptr_t;
  return (unsigned)(unsigned long long)(lds_ptr_t)p;
}

// Async global -> LDS copy, 16 bytes per lane, tracked by ASYNCcnt.
static __device__ __forceinline__ void async_copy16(unsigned lds_off,
                                                    const void* gsrc) {
  asm volatile("global_load_async_to_lds_b128 %0, %1, off"
               :
               : "v"(lds_off), "v"(gsrc)
               : "memory");
}

static __device__ __forceinline__ void wait_async0() {
  asm volatile("s_wait_asynccnt 0x0" ::: "memory");
}

// ---------------------------------------------------------------------------
// Kernel 1: QKV projection.  x[BT,1024] (f32) * W[1024,128] (f32) -> f16.
//   blockIdx.y: 0 -> Q (row major [bt][128])
//               1 -> K (row major [bt][128])
//               2 -> V (transposed  [b][128][4096])
// 256 threads = 8 waves; wave w computes rows [blk*128 + w*16, +16) x all 128 cols.
// ---------------------------------------------------------------------------
__global__ __launch_bounds__(256) void qkv_proj_kernel(
    const float* __restrict__ x, const float* __restrict__ Wq,
    const float* __restrict__ Wk, const float* __restrict__ Wv,
    _Float16* __restrict__ Qh, _Float16* __restrict__ Kh,
    _Float16* __restrict__ Vt) {
  alignas(32) __shared__ _Float16 Wt[DH][CHUNK];  // W chunk transposed, 16 KB

  const int which = blockIdx.y;
  const float* __restrict__ W = (which == 0) ? Wq : ((which == 1) ? Wk : Wv);

  const int tid  = threadIdx.x;
  const int wave = tid >> 5;
  const int lane = tid & 31;
  const int r    = lane & 15;
  const int g    = lane >> 4;
  const int rowBase = blockIdx.x * 128 + wave * 16;

  v8f acc[8];
#pragma unroll
  for (int f = 0; f < 8; ++f) acc[f] = vzero8();

  for (int c0 = 0; c0 < D_IN; c0 += CHUNK) {
    // ---- cooperative transpose-load of W[c0..c0+63][0..127] into LDS (f16)
#pragma unroll
    for (int i = 0; i < 8; ++i) {
      int f4 = tid + 256 * i;        // 0..2047 float4 tiles
      int c  = (f4 * 4) / DH;        // 0..63
      int n  = (f4 * 4) % DH;        // multiple of 4
      v4f w  = *(const v4f*)(W + (size_t)(c0 + c) * DH + n);
      Wt[n + 0][c] = (_Float16)w.x;
      Wt[n + 1][c] = (_Float16)w.y;
      Wt[n + 2][c] = (_Float16)w.z;
      Wt[n + 3][c] = (_Float16)w.w;
    }
    __syncthreads();

#pragma unroll
    for (int kc = 0; kc < CHUNK; kc += 32) {
      // A fragment (16x32) straight from x, f32 -> f16
      const float* xrow = x + (size_t)(rowBase + r) * D_IN + c0 + kc;
      v4f f0 = *(const v4f*)(xrow + g * 16);
      v4f f1 = *(const v4f*)(xrow + g * 16 + 4);
      v4f f2 = *(const v4f*)(xrow + g * 16 + 8);
      v4f f3 = *(const v4f*)(xrow + g * 16 + 12);
      v16h a;
      a[0]  = (_Float16)f0.x; a[1]  = (_Float16)f0.y;
      a[2]  = (_Float16)f0.z; a[3]  = (_Float16)f0.w;
      a[4]  = (_Float16)f1.x; a[5]  = (_Float16)f1.y;
      a[6]  = (_Float16)f1.z; a[7]  = (_Float16)f1.w;
      a[8]  = (_Float16)f2.x; a[9]  = (_Float16)f2.y;
      a[10] = (_Float16)f2.z; a[11] = (_Float16)f2.w;
      a[12] = (_Float16)f3.x; a[13] = (_Float16)f3.y;
      a[14] = (_Float16)f3.z; a[15] = (_Float16)f3.w;

#pragma unroll
      for (int f = 0; f < 8; ++f) {
        int n = f * 16 + r;  // output column this lane owns
        v16h b = *(const v16h*)(&Wt[n][kc + g * 16]);
        acc[f] = wmma16(a, b, acc[f]);
      }
    }
    __syncthreads();
  }

  // ---- store result (f16)
  if (which == 2) {
#pragma unroll
    for (int f = 0; f < 8; ++f) {
      int n = f * 16 + r;
#pragma unroll
      for (int j = 0; j < 8; ++j) {
        int row = rowBase + j + g * 8;
        int b   = row >> 12;       // / SEQ
        int t   = row & (SEQ - 1);
        Vt[(size_t)b * DH * SEQ + (size_t)n * SEQ + t] = (_Float16)acc[f][j];
      }
    }
  } else {
    _Float16* __restrict__ O = (which == 0) ? Qh : Kh;
#pragma unroll
    for (int f = 0; f < 8; ++f) {
      int n = f * 16 + r;
#pragma unroll
      for (int j = 0; j < 8; ++j) {
        int row = rowBase + j + g * 8;
        O[(size_t)row * DH + n] = (_Float16)acc[f][j];
      }
    }
  }
}

// ---------------------------------------------------------------------------
// Kernel 2: causal flash attention with async double-buffered K/V staging.
// grid = (SEQ/64, B), 128 threads = 4 waves; wave w owns 16 query rows.
// Key blocks of 64 streamed global->LDS via global_load_async_to_lds_b128
// (ASYNCcnt) into ping-pong buffers; ONE barrier per key block.
// Reference quirk: softmax on UNSCALED logits, divide by sqrt(d_k) at the end.
// ---------------------------------------------------------------------------
__global__ __launch_bounds__(128) void attn_kernel(
    const _Float16* __restrict__ Qh, const _Float16* __restrict__ Kh,
    const _Float16* __restrict__ Vt, float* __restrict__ out) {
  alignas(32) __shared__ _Float16 Ks[2][64][DH];   // 64 keys x 128 dims, 2x16 KB
  alignas(32) __shared__ _Float16 Vs[2][DH][64];   // 128 vdims x 64 keys, 2x16 KB
  alignas(32) __shared__ _Float16 Ps[4][16][32];   // per-wave P relayout, 4 KB

  const int b    = blockIdx.y;
  const int qb   = blockIdx.x * 64;
  const int tid  = threadIdx.x;
  const int wave = tid >> 5;
  const int lane = tid & 31;
  const int r    = lane & 15;
  const int g    = lane >> 4;
  const int q0   = qb + wave * 16;

  const _Float16* __restrict__ Qb = Qh + (size_t)b * SEQ * DH;
  const _Float16* __restrict__ Kb = Kh + (size_t)b * SEQ * DH;
  const _Float16* __restrict__ Vb = Vt + (size_t)b * DH * SEQ;

  const unsigned ksBase0 = lds_addr(&Ks[0][0][0]);
  const unsigned ksBase1 = lds_addr(&Ks[1][0][0]);
  const unsigned vsBase0 = lds_addr(&Vs[0][0][0]);
  const unsigned vsBase1 = lds_addr(&Vs[1][0][0]);

  // ---- issue async staging of key block s0 into buffer `buf`
  auto stage = [&](int s0, int buf) {
    const unsigned ksB = buf ? ksBase1 : ksBase0;
    const unsigned vsB = buf ? vsBase1 : vsBase0;
    const char* ksrc = (const char*)(Kb + (size_t)s0 * DH);
#pragma unroll
    for (int i = 0; i < 8; ++i) {  // 16 KB: 128 thr * 8 * 16B
      unsigned boff = (unsigned)(tid + 128 * i) * 16u;
      async_copy16(ksB + boff, ksrc + boff);
    }
    const char* vsrc = (const char*)(Vb + s0 + (size_t)tid * SEQ);  // row n=tid
#pragma unroll
    for (int i = 0; i < 8; ++i) {  // 128 B per vdim row
      async_copy16(vsB + (unsigned)tid * 128u + (unsigned)i * 16u,
                   vsrc + i * 16);
    }
  };

  // Q fragments for this wave's 16 rows: 4 chunks over d=128, kept in VGPRs
  v16h aq[4];
  {
    const _Float16* qrow = Qb + (size_t)(q0 + r) * DH;
#pragma unroll
    for (int c = 0; c < 4; ++c) {
      v8h lo = *(const v8h*)(qrow + c * 32 + g * 16);
      v8h hi = *(const v8h*)(qrow + c * 32 + g * 16 + 8);
      v16h a;
#pragma unroll
      for (int i = 0; i < 8; ++i) { a[i] = lo[i]; a[8 + i] = hi[i]; }
      aq[c] = a;
    }
  }

  v8f o[8];
#pragma unroll
  for (int f = 0; f < 8; ++f) o[f] = vzero8();
  float m[8], l[8];
#pragma unroll
  for (int j = 0; j < 8; ++j) { m[j] = -3.0e38f; l[j] = 0.0f; }

  stage(0, 0);  // preload first key block

  int buf = 0;
  for (int s0 = 0; s0 <= qb; s0 += 64) {
    wait_async0();    // my async loads into `buf` are in LDS
    __syncthreads();  // everyone's are; previous readers of buf^1 are done

    if (s0 + 64 <= qb) stage(s0 + 64, buf ^ 1);  // overlap next block with compute

    const bool diag = (s0 == qb);

#pragma unroll
    for (int h = 0; h < 2; ++h) {  // two 32-key halves of this block
      // ---- S = Q K^T for keys [s0+h*32, +32): 2 fragments x 4 d-chunks
      v8f s[2];
#pragma unroll
      for (int kn = 0; kn < 2; ++kn) {
        v8f sa = vzero8();
        int key = h * 32 + kn * 16 + r;  // local key this lane owns
#pragma unroll
        for (int c = 0; c < 4; ++c) {
          v16h bk = *(const v16h*)(&Ks[buf][key][c * 32 + g * 16]);
          sa = wmma16(aq[c], bk, sa);
        }
        if (diag) {
          int ki = s0 + key;
#pragma unroll
          for (int j = 0; j < 8; ++j) {
            int qi = q0 + j + g * 8;
            if (ki > qi) sa[j] = -3.0e38f;
          }
        }
        s[kn] = sa;
      }

      // ---- online softmax update over these 32 keys
      float alpha[8];
#pragma unroll
      for (int j = 0; j < 8; ++j) {
        float v = fmaxf(s[0][j], s[1][j]);
        v = fmaxf(v, __shfl_xor(v, 1));
        v = fmaxf(v, __shfl_xor(v, 2));
        v = fmaxf(v, __shfl_xor(v, 4));
        v = fmaxf(v, __shfl_xor(v, 8));
        float mn = fmaxf(m[j], v);
        alpha[j] = __expf(m[j] - mn);
        m[j] = mn;
      }
#pragma unroll
      for (int j = 0; j < 8; ++j) {
        float p0 = __expf(s[0][j] - m[j]);
        float p1 = __expf(s[1][j] - m[j]);
        Ps[wave][j + g * 8][r]      = (_Float16)p0;
        Ps[wave][j + g * 8][16 + r] = (_Float16)p1;
        float rs = p0 + p1;
        rs += __shfl_xor(rs, 1);
        rs += __shfl_xor(rs, 2);
        rs += __shfl_xor(rs, 4);
        rs += __shfl_xor(rs, 8);
        l[j] = l[j] * alpha[j] + rs;
#pragma unroll
        for (int f = 0; f < 8; ++f) o[f][j] *= alpha[j];
      }

      // wave-private LDS relayout: wait for our DS stores before re-reading
      asm volatile("s_wait_dscnt 0" ::: "memory");

      // ---- A fragment of P (16 q-rows x 32 keys)
      v16h pa;
      {
        const _Float16* prow = &Ps[wave][r][0];
        v8h lo = *(const v8h*)(prow + g * 16);
        v8h hi = *(const v8h*)(prow + g * 16 + 8);
#pragma unroll
        for (int i = 0; i < 8; ++i) { pa[i] = lo[i]; pa[8 + i] = hi[i]; }
      }

      // ---- O += P V  (8 v-dim fragments)
#pragma unroll
      for (int f = 0; f < 8; ++f) {
        v16h bv = *(const v16h*)(&Vs[buf][f * 16 + r][h * 32 + g * 16]);
        o[f] = wmma16(pa, bv, o[f]);
      }
    }

    buf ^= 1;
  }

  // ---- normalize, apply reference's post-AV 1/sqrt(d_k), store f32
  const float scale = 0.08838834764831845f;  // 1/sqrt(128)
#pragma unroll
  for (int j = 0; j < 8; ++j) {
    float inv = scale / l[j];
    int q = q0 + j + g * 8;
    float* orow = out + ((size_t)b * SEQ + q) * DH;
#pragma unroll
    for (int f = 0; f < 8; ++f) {
      orow[f * 16 + r] = o[f][j] * inv;
    }
  }
}

// ---------------------------------------------------------------------------
extern "C" void kernel_launch(void* const* d_in, const int* in_sizes, int n_in,
                              void* d_out, int out_size, void* d_ws,
                              size_t ws_size, hipStream_t stream) {
  (void)in_sizes; (void)n_in; (void)out_size; (void)ws_size;
  const float* x  = (const float*)d_in[0];
  const float* Wq = (const float*)d_in[1];
  const float* Wk = (const float*)d_in[2];
  const float* Wv = (const float*)d_in[3];

  _Float16* Qh = (_Float16*)d_ws;                     // [BT][128] f16, 4 MB
  _Float16* Kh = Qh + (size_t)BT * DH;                // [BT][128] f16, 4 MB
  _Float16* Vt = Kh + (size_t)BT * DH;                // [B][128][SEQ] f16, 4 MB

  qkv_proj_kernel<<<dim3(BT / 128, 3), 256, 0, stream>>>(x, Wq, Wk, Wv, Qh, Kh,
                                                         Vt);
  attn_kernel<<<dim3(SEQ / 64, NBAT), 128, 0, stream>>>(Qh, Kh, Vt,
                                                        (float*)d_out);
}
